// MultiHeadAttention_53498112639051
// MI455X (gfx1250) — compile-verified, matrix operand order checked by reading, not run on hardware
//
#include <hip/hip_runtime.h>
#include <hip/hip_bf16.h>
#include <cstdint>

#define HEADS    16
#define CHANNELS 1024
#define CPH      64
#define BATCH    2
#define SLEN     2048
#define MWORDS   (SLEN / 32)   // packed mask words per query row

typedef __attribute__((ext_vector_type(16))) __bf16 v16bf;
typedef __attribute__((ext_vector_type(8)))  __bf16 v8bf;
typedef __attribute__((ext_vector_type(8)))  float  v8f;
typedef __attribute__((ext_vector_type(4)))  float  v4f;

// 16-bit WMMA fragment layout (ISA 7.12.2): lane owns row (A) / col (B) = lane%16,
// elements 0..7 hold K = half*8+(0..7), elements 8..15 hold K = 16+half*8+(0..7)
// where half = lane/16. Each fragment = two contiguous 16B runs.

__device__ __forceinline__ v16bf frag16(const __bf16* lo_p, const __bf16* hi_p) {
  v8bf lo = *(const v8bf*)lo_p;
  v8bf hi = *(const v8bf*)hi_p;
  return __builtin_shufflevector(lo, hi, 0, 1, 2, 3, 4, 5, 6, 7,
                                 8, 9, 10, 11, 12, 13, 14, 15);
}

// ---------------------------------------------------------------------------
// Mask prepass: pack bool mask [B,S,S] into bitmask words [B,S,S/32].
// One wave packs one 32-key word via ballot.
// ---------------------------------------------------------------------------
__global__ void __launch_bounds__(256)
mask_pack_kernel(const unsigned char* __restrict__ mask,
                 unsigned* __restrict__ mp)
{
  const int w = blockIdx.x * (blockDim.x >> 5) + (threadIdx.x >> 5);
  const int lane = threadIdx.x & 31;
  const unsigned char mv = mask[(size_t)w * 32 + lane];
  const unsigned bits = (unsigned)__ballot(mv != 0);
  if (lane == 0) mp[w] = bits;
}

// ---------------------------------------------------------------------------
// Projection: Y = (X^T W^T + b) * oscale, bf16 out.
// TRANS=0: Y[b,h,s,d]  (Q,K)   TRANS=1: Y[b,h,d,s]  (V, PV B-layout)
// One wave = one 16x16 tile, K-loop of 32 with v_wmma_f32_16x16x32_bf16.
// ---------------------------------------------------------------------------
template <bool TRANS>
__global__ void __launch_bounds__(128)
qkv_proj_kernel(const float* __restrict__ X,    // [B, C, S]
                const float* __restrict__ W,    // [C_out, C_in]
                const float* __restrict__ bias, // [C_out]
                __bf16* __restrict__ Y,
                float oscale)
{
  const int lane = threadIdx.x & 31;
  const int wave = threadIdx.x >> 5;
  const int half = lane >> 4;
  const int l16  = lane & 15;

  const int m0 = blockIdx.x * 16;                    // row tile in (b,s)
  const int n  = (blockIdx.y * 4 + wave) * 16 + l16; // lane's out channel
  const int b  = m0 / SLEN;
  const int s0 = m0 - b * SLEN;

  const float* xcol = X + (size_t)b * CHANNELS * SLEN + (s0 + l16); // A row m=l16
  const float* wrow = W + (size_t)n * CHANNELS;                     // B col n=l16

  v8f acc = (v8f){};
  for (int k0 = 0; k0 < CHANNELS; k0 += 32) {
    if (k0 + 32 < CHANNELS) {
      __builtin_prefetch(&xcol[(size_t)(k0 + 32) * SLEN], 0, 0);
      __builtin_prefetch(&wrow[k0 + 32], 0, 0);
    }
    v8f wlo = *(const v8f*)(wrow + k0 + half * 8);
    v8f whi = *(const v8f*)(wrow + k0 + 16 + half * 8);
    v16bf a, bm;
#pragma unroll
    for (int j = 0; j < 8; ++j) {
      bm[j]     = (__bf16)wlo[j];
      bm[j + 8] = (__bf16)whi[j];
    }
    // A fragment: strided gather (X is [B,C,S]; lanes coalesce along s)
#pragma unroll
    for (int j = 0; j < 8; ++j) {
      a[j]     = (__bf16)xcol[(size_t)(k0 + half * 8 + j) * SLEN];
      a[j + 8] = (__bf16)xcol[(size_t)(k0 + 16 + half * 8 + j) * SLEN];
    }
    acc = __builtin_amdgcn_wmma_f32_16x16x32_bf16(false, a, false, bm,
                                                  (short)0, acc, false, false);
  }

  const int h = n / CPH, d = n % CPH;
  const float bval = bias[n];
  if (TRANS) {
    v8bf pack;                                  // D rows contiguous in s
#pragma unroll
    for (int r = 0; r < 8; ++r) pack[r] = (__bf16)((acc[r] + bval) * oscale);
    __bf16* dst = Y + (((size_t)b * HEADS + h) * CPH + d) * SLEN + s0 + 8 * half;
    *(v8bf*)dst = pack;
  } else {
    __bf16* yrow = Y + (((size_t)b * HEADS + h) * SLEN) * CPH + d;
#pragma unroll
    for (int r = 0; r < 8; ++r) {
      const int s = s0 + r + 8 * half;
      yrow[(size_t)s * CPH] = (__bf16)((acc[r] + bval) * oscale);
    }
  }
}

// ---------------------------------------------------------------------------
// Flash attention. Block = 4 waves sharing one (b,h); each wave = 16 queries.
// K/V^T tiles async-staged to LDS, double-buffered. Softmax done in A-layout
// (lane owns one score row) so the exp'd values directly form the PV A-frag.
// ---------------------------------------------------------------------------
__global__ void __launch_bounds__(128)
attn_kernel(const __bf16* __restrict__ qh,   // [B,H,S,CPH] (pre-scaled by 1/8)
            const __bf16* __restrict__ kh,   // [B,H,S,CPH]
            const __bf16* __restrict__ vt,   // [B,H,CPH,S] (transposed V)
            const unsigned* __restrict__ mp, // [B,S,S/32] packed mask bits
            float* __restrict__ out)         // [B,S,C] fp32
{
  __shared__ __align__(16) __bf16 Kt[2][32][CPH]; // 2 x 4KB key tiles
  __shared__ __align__(16) __bf16 Vt[2][CPH][32]; // 2 x 4KB V^T tiles
  __shared__ __align__(16) float  Sl[4][16][32];  // per-wave raw score tiles

  const int tid  = threadIdx.x;
  const int lane = tid & 31;
  const int wave = tid >> 5;
  const int half = lane >> 4;
  const int l16  = lane & 15;

  const int b  = blockIdx.z;
  const int h  = blockIdx.y;
  const int q0 = (blockIdx.x * 4 + wave) * 16;

  const size_t headBase = ((size_t)b * HEADS + h) * SLEN; // q/k row base
  const size_t vBase    = ((size_t)b * HEADS + h) * CPH;  // vt row base

  auto stage = [&](int kb, int buf) {                 // async K/V^T staging
#pragma unroll
    for (int c = tid; c < 256; c += 128) {            // K tile: 32 x 128B
      const int row = c >> 3, col = (c & 7) * 8;
      unsigned lds = (unsigned)(uintptr_t)&Kt[buf][row][col];
      unsigned long long ga =
          (unsigned long long)(uintptr_t)(kh + (headBase + kb + row) * CPH + col);
      asm volatile("global_load_async_to_lds_b128 %0, %1, off"
                   :: "v"(lds), "v"(ga) : "memory");
    }
#pragma unroll
    for (int c = tid; c < 256; c += 128) {            // V^T tile: 64 x 64B
      const int d = c >> 2, col = (c & 3) * 8;
      unsigned lds = (unsigned)(uintptr_t)&Vt[buf][d][col];
      unsigned long long ga =
          (unsigned long long)(uintptr_t)(vt + (vBase + d) * SLEN + kb + col);
      asm volatile("global_load_async_to_lds_b128 %0, %1, off"
                   :: "v"(lds), "v"(ga) : "memory");
    }
  };

  // Q tile (16x64): two A fragments (Q already carries the 1/sqrt(64) scale)
  v16bf aq[2];
  {
    const __bf16* qrow = qh + (headBase + q0 + l16) * CPH;
#pragma unroll
    for (int c = 0; c < 2; ++c)
      aq[c] = frag16(qrow + c * 32 + half * 8, qrow + c * 32 + 16 + half * 8);
  }

  float m_run = -1e30f, l_run = 0.0f;   // stats for this lane's row (q0+l16)
  v8f o[4];
#pragma unroll
  for (int t = 0; t < 4; ++t) o[t] = (v8f){};

  const unsigned* mrow = mp + ((size_t)b * SLEN + q0 + l16) * MWORDS;
  const int nBlk = SLEN / 32;

  stage(0, 0);
  for (int i = 0; i < nBlk; ++i) {
    const int buf = i & 1;
    if (i + 1 < nBlk) {
      stage((i + 1) * 32, buf ^ 1);                   // overlap next tile
      asm volatile("s_wait_asynccnt 0x4" ::: "memory");
    } else {
      asm volatile("s_wait_asynccnt 0x0" ::: "memory");
    }
    __syncthreads();

    // ---- scores: two 16x16 tiles, K-depth 64 ----
    v8f st[2];
#pragma unroll
    for (int t = 0; t < 2; ++t) {
      const __bf16* krow = &Kt[buf][16 * t + l16][0]; // B col n = key
      v8f acc = (v8f){};
      acc = __builtin_amdgcn_wmma_f32_16x16x32_bf16(
          false, aq[0], false, frag16(krow + half * 8, krow + 16 + half * 8),
          (short)0, acc, false, false);
      acc = __builtin_amdgcn_wmma_f32_16x16x32_bf16(
          false, aq[1], false, frag16(krow + 32 + half * 8, krow + 48 + half * 8),
          (short)0, acc, false, false);
      st[t] = acc;
    }

    // ---- D-layout -> LDS (raw scores) ----
#pragma unroll
    for (int r = 0; r < 8; ++r) {
      Sl[wave][r + 8 * half][l16]      = st[0][r];
      Sl[wave][r + 8 * half][16 + l16] = st[1][r];
    }

    // ---- A-layout reload: this lane's 16 fragment elements of row l16 ----
    const float* srow = &Sl[wave][l16][0];
    v4f a0 = *(const v4f*)(srow + half * 8);
    v4f a1 = *(const v4f*)(srow + half * 8 + 4);
    v4f b0 = *(const v4f*)(srow + 16 + half * 8);
    v4f b1 = *(const v4f*)(srow + 16 + half * 8 + 4);
    float v[16];
#pragma unroll
    for (int j = 0; j < 4; ++j) {
      v[j] = a0[j]; v[4 + j] = a1[j]; v[8 + j] = b0[j]; v[12 + j] = b1[j];
    }

    // ---- mask: one packed word covers all 32 keys of this row ----
    const unsigned mw  = mrow[i];
    const unsigned mlo = (mw >> (half * 8)) & 0xffu;
    const unsigned mhi = (mw >> (16 + half * 8)) & 0xffu;
#pragma unroll
    for (int j = 0; j < 8; ++j) {
      if ((mlo >> j) & 1u) v[j]     = -1e9f;
      if ((mhi >> j) & 1u) v[8 + j] = -1e9f;
    }

    // ---- online softmax: local 16-way + one cross-half shuffle ----
    float mx = v[0];
#pragma unroll
    for (int j = 1; j < 16; ++j) mx = fmaxf(mx, v[j]);
    mx = fmaxf(mx, __shfl_xor(mx, 16, 32));
    const float mn    = fmaxf(m_run, mx);
    const float alpha = __expf(m_run - mn);
    float ps = 0.0f;
#pragma unroll
    for (int j = 0; j < 16; ++j) { v[j] = __expf(v[j] - mn); ps += v[j]; }
    ps += __shfl_xor(ps, 16, 32);
    l_run = l_run * alpha + ps;
    m_run = mn;

    // exp'd values ARE the PV A-fragment
    v16bf ap;
#pragma unroll
    for (int j = 0; j < 16; ++j) ap[j] = (__bf16)v[j];

    // ---- broadcast alpha to D-layout rows, rescale ctx ----
#pragma unroll
    for (int t = 0; t < 4; ++t)
#pragma unroll
      for (int r = 0; r < 8; ++r)
        o[t][r] *= __shfl(alpha, r + 8 * half, 32);

    // ---- ctx += P (16x32) x V (32x64): 4 WMMAs over d tiles ----
#pragma unroll
    for (int t = 0; t < 4; ++t) {
      const __bf16* vrow = &Vt[buf][16 * t + l16][0]; // B col n = d
      o[t] = __builtin_amdgcn_wmma_f32_16x16x32_bf16(
          false, ap, false, frag16(vrow + half * 8, vrow + 16 + half * 8),
          (short)0, o[t], false, false);
    }

    __syncthreads();                                  // done with this buffer
  }

  // ---- normalize (gather l per D-row) and store: out[b, q, h*64+d] ----
  float l_d[8];
#pragma unroll
  for (int r = 0; r < 8; ++r) l_d[r] = __shfl(l_run, r + 8 * half, 32);
#pragma unroll
  for (int t = 0; t < 4; ++t) {
    const int d = h * CPH + 16 * t + l16;
#pragma unroll
    for (int r = 0; r < 8; ++r) {
      const int row = q0 + r + 8 * half;
      out[((size_t)b * SLEN + row) * CHANNELS + d] = o[t][r] / l_d[r];
    }
  }
}

// ---------------------------------------------------------------------------
extern "C" void kernel_launch(void* const* d_in, const int* in_sizes, int n_in,
                              void* d_out, int out_size, void* d_ws, size_t ws_size,
                              hipStream_t stream) {
  const float* Q  = (const float*)d_in[0];
  const float* K  = (const float*)d_in[1];
  const float* V  = (const float*)d_in[2];
  const unsigned char* mask = (const unsigned char*)d_in[3];
  const float* Wq = (const float*)d_in[4];
  const float* bq = (const float*)d_in[5];
  const float* Wk = (const float*)d_in[6];
  const float* bk = (const float*)d_in[7];
  const float* Wv = (const float*)d_in[8];
  const float* bv = (const float*)d_in[9];
  float* out = (float*)d_out;

  const size_t nPerTensor = (size_t)BATCH * HEADS * SLEN * CPH; // 4M elems
  __bf16* qh = (__bf16*)d_ws;               // 3 x 8 MB bf16 tensors
  __bf16* kh = qh + nPerTensor;
  __bf16* vh = kh + nPerTensor;             // transposed [B,H,CPH,S]
  unsigned* mp = (unsigned*)(vh + nPerTensor); // 1 MB packed mask

  const int nWords = BATCH * SLEN * MWORDS;            // 262144
  mask_pack_kernel<<<nWords / 8, 256, 0, stream>>>(mask, mp);

  dim3 pgrid(BATCH * SLEN / 16, CHANNELS / 64, 1);     // 256 x 16
  qkv_proj_kernel<false><<<pgrid, 128, 0, stream>>>(Q, Wq, bq, qh, 0.125f);
  qkv_proj_kernel<false><<<pgrid, 128, 0, stream>>>(K, Wk, bk, kh, 1.0f);
  qkv_proj_kernel<true ><<<pgrid, 128, 0, stream>>>(V, Wv, bv, vh, 1.0f);

  dim3 agrid(SLEN / 64, HEADS, BATCH);                 // 32 x 16 x 2
  attn_kernel<<<agrid, 128, 0, stream>>>(qh, kh, vh, mp, out);
}
